// RoPEViT_8134668058985
// MI455X (gfx1250) — compile-verified, matrix operand order checked by reading, not run on hardware
//
#include <hip/hip_runtime.h>
#include <hip/hip_bf16.h>

typedef __attribute__((ext_vector_type(16))) __bf16 v16bf;
typedef __attribute__((ext_vector_type(8)))  float  v8f;

#define BV   32
#define LV   12
#define DV   768
#define FFV  3072
#define HV   12
#define HDV  64
#define NV   197   // tokens (196 patches + cls)
#define NPV  196
#define NCV  1000

// ---------------------------------------------------------------------------
// WMMA helpers
// ---------------------------------------------------------------------------
__device__ __forceinline__ v8f wmma_bf16(v16bf a, v16bf b, v8f c) {
  return __builtin_amdgcn_wmma_f32_16x16x32_bf16(false, a, false, b,
                                                 (short)0, c, false, false);
}

// Load a 16x32 bf16 A/B fragment from an LDS (or global) tile.
// Per ISA 16-bit layout: lanes 0-15 -> row, K = {0..7,16..23}; lanes 16-31 ->
// same rows, K = {8..15,24..31}. Two 16B reads per lane.
__device__ __forceinline__ v16bf ld_frag(const __bf16* p, int row, int stride) {
  int lane = threadIdx.x & 31;
  const __bf16* rp = p + (size_t)(row + (lane & 15)) * stride + ((lane & 16) ? 8 : 0);
  union { uint4 u[2]; v16bf v; } f;
  f.u[0] = *(const uint4*)rp;
  f.u[1] = *(const uint4*)(rp + 16);
  return f.v;
}

__device__ __forceinline__ v16bf ld_frag_clamp(const __bf16* p, int row, int maxrow,
                                               int stride, int koff) {
  int lane = threadIdx.x & 31;
  int r = row + (lane & 15);
  if (r >= maxrow) r = maxrow - 1;
  const __bf16* rp = p + (size_t)r * stride + koff + ((lane & 16) ? 8 : 0);
  union { uint4 u[2]; v16bf v; } f;
  f.u[0] = *(const uint4*)rp;
  f.u[1] = *(const uint4*)(rp + 16);
  return f.v;
}

// ---------------------------------------------------------------------------
// Generic WMMA GEMM: C[M,N] = A[M,K](bf16, row-major) * W[N,K](fp32)^T
// Software-pipelined: next K-tile is prefetched into registers while the
// current tile's 8 WMMAs execute. Weight fp32->bf16 conversion happens at
// LDS-commit time with packed 8-byte stores.
// Epilogue: optional bias, exact GELU, residual add, fp32 or bf16 output.
// RowMap==1 remaps output row m -> (m/196)*197 + (m%196) + 1 (patch embed).
// ---------------------------------------------------------------------------
template <bool HasBias, bool Gelu, bool Residual, bool OutBf16, int RowMap>
__global__ __launch_bounds__(256) void gemm_bf16_wmma(
    const __bf16* __restrict__ A, const float* __restrict__ Bw,
    const float* __restrict__ bias, const float* __restrict__ resid,
    float* __restrict__ outF, __bf16* __restrict__ outB,
    int M, int N, int K) {
  __shared__ __bf16 Ash[128 * 40];
  __shared__ __bf16 Bsh[128 * 40];
  const int t = threadIdx.x;
  const int mBase = blockIdx.y * 128;
  const int nBase = blockIdx.x * 128;
  const int wave = t >> 5, lane = t & 31;
  const int wm = (wave & 1) * 64;   // 2 wave rows
  const int wn = (wave >> 1) * 32;  // 4 wave cols

  v8f acc[4][2];
#pragma unroll
  for (int i = 0; i < 4; ++i)
#pragma unroll
    for (int j = 0; j < 2; ++j) { v8f z = {}; acc[i][j] = z; }

  uint4 aReg[2];
  float4 bReg[4];
  auto prefetch = [&](int k0) {
#pragma unroll
    for (int s = 0; s < 2; ++s) {               // A: 512 chunks of 8 bf16
      int c = t * 2 + s;
      int row = c >> 2, coff = (c & 3) * 8;
      int gr = mBase + row;
      aReg[s] = make_uint4(0u, 0u, 0u, 0u);
      if (gr < M) aReg[s] = *(const uint4*)(A + (size_t)gr * K + k0 + coff);
    }
#pragma unroll
    for (int s = 0; s < 4; ++s) {               // W: 1024 chunks of 4 fp32
      int c = t * 4 + s;
      int row = c >> 3, coff = (c & 7) * 4;
      int gr = nBase + row;
      bReg[s] = make_float4(0.f, 0.f, 0.f, 0.f);
      if (gr < N) bReg[s] = *(const float4*)(Bw + (size_t)gr * K + k0 + coff);
    }
  };

  prefetch(0);
  for (int k0 = 0; k0 < K; k0 += 32) {
    __syncthreads();  // previous iteration done reading LDS
    // Commit staged registers to LDS.
#pragma unroll
    for (int s = 0; s < 2; ++s) {
      int c = t * 2 + s;
      int row = c >> 2, coff = (c & 3) * 8;
      *(uint4*)(&Ash[row * 40 + coff]) = aReg[s];
    }
#pragma unroll
    for (int s = 0; s < 4; ++s) {
      int c = t * 4 + s;
      int row = c >> 3, coff = (c & 7) * 4;
      union { __bf16 h[4]; uint2 u; } pk;
      pk.h[0] = (__bf16)bReg[s].x; pk.h[1] = (__bf16)bReg[s].y;
      pk.h[2] = (__bf16)bReg[s].z; pk.h[3] = (__bf16)bReg[s].w;
      *(uint2*)(&Bsh[row * 40 + coff]) = pk.u;  // one 8B store per chunk
    }
    if (k0 + 32 < K) prefetch(k0 + 32);         // overlap next loads w/ WMMA
    __syncthreads();

    v16bf af[4], bfr[2];
#pragma unroll
    for (int i = 0; i < 4; ++i) af[i] = ld_frag(Ash, wm + i * 16, 40);
#pragma unroll
    for (int j = 0; j < 2; ++j) bfr[j] = ld_frag(Bsh, wn + j * 16, 40);
#pragma unroll
    for (int i = 0; i < 4; ++i)
#pragma unroll
      for (int j = 0; j < 2; ++j)
        acc[i][j] = wmma_bf16(af[i], bfr[j], acc[i][j]);
  }

  // Epilogue. C layout: lanes 0-15 -> (M = vv, N = lane); lanes 16-31 -> M = 8+vv.
  const int cloc = lane & 15;
#pragma unroll
  for (int i = 0; i < 4; ++i)
#pragma unroll
    for (int j = 0; j < 2; ++j)
#pragma unroll
      for (int vv = 0; vv < 8; ++vv) {
        int m = mBase + wm + i * 16 + ((lane < 16) ? vv : 8 + vv);
        int n = nBase + wn + j * 16 + cloc;
        if (m < M && n < N) {
          float val = acc[i][j][vv];
          if (HasBias) val += bias[n];
          if (Gelu) val = 0.5f * val * (1.0f + erff(val * 0.70710678118654752f));
          size_t orow = (RowMap == 1)
                            ? (size_t)((m / 196) * 197 + (m % 196) + 1)
                            : (size_t)m;
          size_t oidx = orow * (size_t)N + n;
          if (Residual) val += resid[oidx];
          if (OutBf16) outB[oidx] = (__bf16)val; else outF[oidx] = val;
        }
      }
}

// ---------------------------------------------------------------------------
// LayerNorm (fp32 in, bf16 out). One wave per token, D=768 -> 24 elems/lane.
// ---------------------------------------------------------------------------
__global__ __launch_bounds__(256) void layernorm_to_bf16(
    const float* __restrict__ x, const float* __restrict__ w,
    const float* __restrict__ b, __bf16* __restrict__ out, int T) {
  int wave = threadIdx.x >> 5, lane = threadIdx.x & 31;
  int tok = blockIdx.x * 8 + wave;
  if (tok >= T) return;
  const float* xr = x + (size_t)tok * DV;
  float s = 0.f, sq = 0.f;
  float vals[24];
#pragma unroll
  for (int i = 0; i < 24; ++i) {
    float xv = xr[lane + 32 * i];
    vals[i] = xv; s += xv; sq += xv * xv;
  }
#pragma unroll
  for (int m = 1; m < 32; m <<= 1) {
    s += __shfl_xor(s, m);
    sq += __shfl_xor(sq, m);
  }
  float mean = s * (1.0f / DV);
  float var = sq * (1.0f / DV) - mean * mean;
  float rstd = rsqrtf(var + 1e-5f);
  __bf16* orow = out + (size_t)tok * DV;
#pragma unroll
  for (int i = 0; i < 24; ++i) {
    int d = lane + 32 * i;
    orow[d] = (__bf16)((vals[i] - mean) * rstd * w[d] + b[d]);
  }
}

// ---------------------------------------------------------------------------
// RoPE + QKV split: qkv (B*N, 3*D) bf16 -> q,k,v (B,H,N,hd) bf16.
// Interleaved-pair rotation; hd^-0.5 folded into q.
// ---------------------------------------------------------------------------
__global__ void rope_split_kernel(const __bf16* __restrict__ qkv,
                                  __bf16* __restrict__ q, __bf16* __restrict__ k,
                                  __bf16* __restrict__ v) {
  int tid = blockIdx.x * blockDim.x + threadIdx.x;
  const int total = BV * HV * NV * 32;
  if (tid >= total) return;
  int i = tid & 31;
  int n = (tid >> 5) % NV;
  int bh = tid / (32 * NV);
  int b = bh / HV, h = bh % HV;
  float invf = __powf(10000.0f, -(float)(2 * i) / (float)HDV);
  float ang = (float)n * invf;
  float sn = __sinf(ang), cs = __cosf(ang);
  size_t rbase = ((size_t)(b * NV + n)) * (3 * DV) + h * HDV + 2 * i;
  float q1 = (float)qkv[rbase],        q2 = (float)qkv[rbase + 1];
  float k1 = (float)qkv[rbase + DV],   k2 = (float)qkv[rbase + DV + 1];
  float v1 = (float)qkv[rbase + 2*DV], v2 = (float)qkv[rbase + 2*DV + 1];
  size_t obase = (((size_t)bh) * NV + n) * HDV + 2 * i;
  const float scale = 0.125f;  // 64^-0.5
  q[obase]     = (__bf16)((q1 * cs - q2 * sn) * scale);
  q[obase + 1] = (__bf16)((q2 * cs + q1 * sn) * scale);
  k[obase]     = (__bf16)(k1 * cs - k2 * sn);
  k[obase + 1] = (__bf16)(k2 * cs + k1 * sn);
  v[obase]     = (__bf16)v1;
  v[obase + 1] = (__bf16)v2;
}

// ---------------------------------------------------------------------------
// Attention: one block per (b,h). K (208x64) and V^T (64x224) staged in LDS.
// Each wave owns 16-row Q tiles: S=Q*K^T via WMMA, fp32 softmax with
// half-wave shfl reductions, per-wave LDS bounce to re-layout P as an
// A-fragment, O=P*V^T via WMMA.
// ---------------------------------------------------------------------------
__global__ __launch_bounds__(256) void attn_kernel(
    const __bf16* __restrict__ q, const __bf16* __restrict__ k,
    const __bf16* __restrict__ v, __bf16* __restrict__ o) {
  __shared__ __bf16 Ksh[208 * 64];      // [n][d], rows >=197 zero
  __shared__ __bf16 Vsh[64 * 224];      // [d][m] (transposed), m >=197 zero
  __shared__ __bf16 Psh[8][16 * 32];    // per-wave P chunk scratch

  const int bh = blockIdx.x;
  const int b = bh / HV, h = bh % HV;
  const __bf16* qb = q + (size_t)bh * NV * HDV;
  const __bf16* kb = k + (size_t)bh * NV * HDV;
  const __bf16* vb = v + (size_t)bh * NV * HDV;
  const int t = threadIdx.x;

  for (int i = t; i < 208 * 64; i += 256) {
    int n = i >> 6;
    Ksh[i] = (n < NV) ? kb[i] : (__bf16)0.f;
  }
  for (int i = t; i < 64 * 224; i += 256) {
    int d = i / 224, m = i % 224;
    Vsh[i] = (m < NV) ? vb[m * HDV + d] : (__bf16)0.f;
  }
  __syncthreads();

  const int wave = t >> 5, lane = t & 31;
  const int cloc = lane & 15;

  for (int qt = wave; qt < 13; qt += 8) {  // 13 tiles cover 197 rows
    int mq = qt * 16;
    v16bf qa0 = ld_frag_clamp(qb, mq, NV, HDV, 0);
    v16bf qa1 = ld_frag_clamp(qb, mq, NV, HDV, 32);

    v8f S[13];
#pragma unroll
    for (int nt = 0; nt < 13; ++nt) {
      v8f a = {};
      a = wmma_bf16(qa0, ld_frag(Ksh, nt * 16, 64), a);
      a = wmma_bf16(qa1, ld_frag(Ksh + 32, nt * 16, 64), a);
      S[nt] = a;
    }

    float rinv[8];
#pragma unroll
    for (int j = 0; j < 8; ++j) {
      float mx = -3.0e38f;
#pragma unroll
      for (int nt = 0; nt < 13; ++nt) {
        int col = nt * 16 + cloc;
        if (col < NV) mx = fmaxf(mx, S[nt][j]);
      }
      mx = fmaxf(mx, __shfl_xor(mx, 1));
      mx = fmaxf(mx, __shfl_xor(mx, 2));
      mx = fmaxf(mx, __shfl_xor(mx, 4));
      mx = fmaxf(mx, __shfl_xor(mx, 8));
      float sum = 0.f;
#pragma unroll
      for (int nt = 0; nt < 13; ++nt) {
        int col = nt * 16 + cloc;
        float p = (col < NV) ? __expf(S[nt][j] - mx) : 0.f;
        S[nt][j] = p;
        sum += p;
      }
      sum += __shfl_xor(sum, 1);
      sum += __shfl_xor(sum, 2);
      sum += __shfl_xor(sum, 4);
      sum += __shfl_xor(sum, 8);
      rinv[j] = 1.0f / sum;
    }

    v8f O[4];
#pragma unroll
    for (int ot = 0; ot < 4; ++ot) { v8f z = {}; O[ot] = z; }
    __bf16* pw = Psh[wave];
#pragma unroll
    for (int kk = 0; kk < 7; ++kk) {  // 7 chunks of 32 cols (padded to 224)
#pragma unroll
      for (int j = 0; j < 8; ++j) {
        int ml = (lane < 16) ? j : (8 + j);
        int t0 = 2 * kk, t1 = 2 * kk + 1;
        float p0 = S[t0][j] * rinv[j];
        float p1 = (t1 < 13) ? S[t1][j] * rinv[j] : 0.f;
        pw[ml * 32 + cloc] = (__bf16)p0;
        pw[ml * 32 + cloc + 16] = (__bf16)p1;
      }
      asm volatile("s_wait_dscnt 0" ::: "memory");  // wave-local LDS RAW
      v16bf pa = ld_frag(pw, 0, 32);
#pragma unroll
      for (int ot = 0; ot < 4; ++ot)
        O[ot] = wmma_bf16(pa, ld_frag(Vsh + kk * 32, ot * 16, 224), O[ot]);
    }

#pragma unroll
    for (int ot = 0; ot < 4; ++ot)
#pragma unroll
      for (int j = 0; j < 8; ++j) {
        int m = mq + ((lane < 16) ? j : 8 + j);
        if (m < NV) {
          int d = ot * 16 + cloc;
          o[((size_t)b * NV + m) * DV + h * HDV + d] = (__bf16)O[ot][j];
        }
      }
  }
}

// ---------------------------------------------------------------------------
// Small helper kernels
// ---------------------------------------------------------------------------
__global__ void im2col_kernel(const float* __restrict__ x, __bf16* __restrict__ out) {
  int tid = blockIdx.x * blockDim.x + threadIdx.x;
  const int total = BV * NPV * DV;
  if (tid >= total) return;
  int col = tid % DV;
  int p = (tid / DV) % NPV;
  int b = tid / (DV * NPV);
  int c = col >> 8, r = (col >> 4) & 15, cc = col & 15;
  int py = p / 14, px = p % 14;
  out[tid] = (__bf16)x[((size_t)(b * 3 + c) * 224 + py * 16 + r) * 224 + px * 16 + cc];
}

__global__ void write_cls_rows(const float* __restrict__ cls, float* __restrict__ h) {
  int tid = blockIdx.x * blockDim.x + threadIdx.x;
  if (tid >= BV * DV) return;
  int d = tid % DV, b = tid / DV;
  h[(size_t)(b * NV) * DV + d] = cls[d];
}

__global__ void gather_cls_bf16(const __bf16* __restrict__ hn, __bf16* __restrict__ out) {
  int tid = blockIdx.x * blockDim.x + threadIdx.x;
  if (tid >= BV * DV) return;
  int d = tid % DV, b = tid / DV;
  out[tid] = hn[(size_t)(b * NV) * DV + d];
}

// ---------------------------------------------------------------------------
// Host orchestration
// ---------------------------------------------------------------------------
extern "C" void kernel_launch(void* const* d_in, const int* in_sizes, int n_in,
                              void* d_out, int out_size, void* d_ws, size_t ws_size,
                              hipStream_t stream) {
  (void)in_sizes; (void)n_in; (void)out_size; (void)ws_size;
  const float* x        = (const float*)d_in[0];
  const float* conv_w   = (const float*)d_in[1];   // (768, 3*16*16=768) flat
  const float* conv_b   = (const float*)d_in[2];
  const float* cls_tok  = (const float*)d_in[3];
  const float* qkv_w    = (const float*)d_in[4];   // (12, 2304, 768)
  const float* proj_w   = (const float*)d_in[5];   // (12, 768, 768)
  const float* ln1_w    = (const float*)d_in[6];
  const float* ln1_b    = (const float*)d_in[7];
  const float* ln2_w    = (const float*)d_in[8];
  const float* ln2_b    = (const float*)d_in[9];
  const float* ff1_w    = (const float*)d_in[10];  // (12, 3072, 768)
  const float* ff1_b    = (const float*)d_in[11];
  const float* ff2_w    = (const float*)d_in[12];  // (12, 768, 3072)
  const float* ff2_b    = (const float*)d_in[13];
  const float* norm_w   = (const float*)d_in[14];
  const float* norm_b   = (const float*)d_in[15];
  const float* head_w   = (const float*)d_in[16];  // (1000, 768)
  const float* head_b   = (const float*)d_in[17];
  float* logits = (float*)d_out;

  const int BN = BV * NV;       // 6304
  const int BP = BV * NPV;      // 6272

  char* p = (char*)d_ws;
  auto carve = [&](size_t bytes) {
    void* r = (void*)p;
    p += (bytes + 255) & ~(size_t)255;
    return r;
  };
  float*  hbuf = (float*)carve((size_t)BN * DV * 4);
  __bf16* hn   = (__bf16*)carve((size_t)BN * DV * 2);
  __bf16* qkvb = (__bf16*)carve((size_t)BN * 3 * DV * 2);
  __bf16* qb   = (__bf16*)carve((size_t)BV * HV * NV * HDV * 2);
  __bf16* kb   = (__bf16*)carve((size_t)BV * HV * NV * HDV * 2);
  __bf16* vb   = (__bf16*)carve((size_t)BV * HV * NV * HDV * 2);
  __bf16* ob   = (__bf16*)carve((size_t)BN * DV * 2);
  __bf16* ffb  = (__bf16*)carve((size_t)BN * FFV * 2);
  __bf16* im2  = (__bf16*)carve((size_t)BP * DV * 2);
  __bf16* clsg = (__bf16*)carve((size_t)BV * DV * 2);

  const dim3 blk(256);
  const int gM  = (BN + 127) / 128;   // 50
  const int gMp = (BP + 127) / 128;   // 49
  const int lnGrid = (BN + 7) / 8;

  // Patch embed: im2col -> GEMM (bias, row-remap into h), cls row write.
  im2col_kernel<<<(BP * DV + 255) / 256, blk, 0, stream>>>(x, im2);
  write_cls_rows<<<(BV * DV + 255) / 256, blk, 0, stream>>>(cls_tok, hbuf);
  gemm_bf16_wmma<true, false, false, false, 1>
      <<<dim3(DV / 128, gMp), blk, 0, stream>>>(
          im2, conv_w, conv_b, nullptr, hbuf, nullptr, BP, DV, DV);

  for (int l = 0; l < LV; ++l) {
    layernorm_to_bf16<<<lnGrid, blk, 0, stream>>>(
        hbuf, ln1_w + l * DV, ln1_b + l * DV, hn, BN);
    gemm_bf16_wmma<false, false, false, true, 0>
        <<<dim3((3 * DV) / 128, gM), blk, 0, stream>>>(
            hn, qkv_w + (size_t)l * 3 * DV * DV, nullptr, nullptr,
            nullptr, qkvb, BN, 3 * DV, DV);
    rope_split_kernel<<<(BV * HV * NV * 32 + 255) / 256, blk, 0, stream>>>(
        qkvb, qb, kb, vb);
    attn_kernel<<<BV * HV, blk, 0, stream>>>(qb, kb, vb, ob);
    gemm_bf16_wmma<false, false, true, false, 0>
        <<<dim3(DV / 128, gM), blk, 0, stream>>>(
            ob, proj_w + (size_t)l * DV * DV, nullptr, hbuf,
            hbuf, nullptr, BN, DV, DV);
    layernorm_to_bf16<<<lnGrid, blk, 0, stream>>>(
        hbuf, ln2_w + l * DV, ln2_b + l * DV, hn, BN);
    gemm_bf16_wmma<true, true, false, true, 0>
        <<<dim3(FFV / 128, gM), blk, 0, stream>>>(
            hn, ff1_w + (size_t)l * FFV * DV, ff1_b + l * FFV, nullptr,
            nullptr, ffb, BN, FFV, DV);
    gemm_bf16_wmma<true, false, true, false, 0>
        <<<dim3(DV / 128, gM), blk, 0, stream>>>(
            ffb, ff2_w + (size_t)l * DV * FFV, ff2_b + l * DV, hbuf,
            hbuf, nullptr, BN, DV, FFV);
  }

  layernorm_to_bf16<<<lnGrid, blk, 0, stream>>>(hbuf, norm_w, norm_b, hn, BN);
  gather_cls_bf16<<<(BV * DV + 255) / 256, blk, 0, stream>>>(hn, clsg);
  gemm_bf16_wmma<true, false, false, false, 0>
      <<<dim3((NCV + 127) / 128, 1), blk, 0, stream>>>(
          clsg, head_w, head_b, nullptr, logits, nullptr, BV, NCV, DV);
}
